// VAEDecoder_72559177499302
// MI455X (gfx1250) — matrix operand order repaired and checked by reading, not compiled, not run on hardware
//
#include <hip/hip_runtime.h>
#include <hip/hip_bf16.h>
#include <math.h>

// ---------------------------------------------------------------------------
// VAE GRU decoder for MI455X (gfx1250, wave32, WMMA).
//   scores/logits [T,B,V], hidden_f [2,B,H], output [B,H]
// Strategy:
//   * bf16 WMMA (v_wmma_f32_16x16x32_bf16, f32 accumulate) for every GEMM.
//   * Time-invariant pieces hoisted: syn_z gate constant, emb gate GEMM for
//     all timesteps, pred matrix [h1 | sem_z | emb] assembled incrementally.
//   * 128 sequential steps -> 2 small WMMA kernels each (layer0, layer1).
//   * Output projection (85% of FLOPs): block-tiled GEMM. Each block stages a
//     64-col x 64-k chunk of W_out into LDS with GLOBAL_LOAD_ASYNC_TO_LDS_B128
//     (ASYNCcnt path, no VGPR staging) and reuses it across 16 M-tiles
//     (256 pred rows): W_out HBM traffic ~2.6 GB (~112 us @ 23.3 TB/s),
//     balanced vs bf16 WMMA compute.
// ---------------------------------------------------------------------------

typedef __bf16 bf16;
typedef __attribute__((ext_vector_type(16))) __bf16 v16bf;
typedef __attribute__((ext_vector_type(8)))  float  v8f;

#define BB   16          // batch
#define TT   128         // timesteps
#define HH   1024        // hidden
#define GG   3072        // 3*H (gates)
#define EE   512         // embedding
#define VV   32000       // vocab
#define PREDW 2560       // H + LAT + E
#define ROWS (TT * BB)   // 2048

union Frag {
  v16bf v;
  bf16  h[16];
  uint4 q[2];
};

// A-matrix 16x32 bf16 fragment: lane L -> m = L%16, hi = L/16.
// element e : K = k0 + 16*hi + e  (16 consecutive K per lane).
__device__ __forceinline__ v16bf load_a_frag(const bf16* row, int k0, int hi) {
  Frag f;
  f.q[0] = *reinterpret_cast<const uint4*>(row + k0 + 16 * hi);
  f.q[1] = *reinterpret_cast<const uint4*>(row + k0 + 16 * hi + 8);
  return f.v;
}
// B-matrix 32x16 bf16 fragment: lane L -> n = L%16, hi = L/16.
// element e : K = k0 + 16*hi + e  (16 consecutive K per lane).
// Weights are PyTorch [N,K] row-major, so a B column is a weight row.
__device__ __forceinline__ v16bf load_b_frag(const bf16* row, int k0, int hi) {
  Frag f;
  f.q[0] = *reinterpret_cast<const uint4*>(row + k0 + 16 * hi);
  f.q[1] = *reinterpret_cast<const uint4*>(row + k0 + 16 * hi + 8);
  return f.v;
}

__device__ __forceinline__ v8f wmma_bf16(v16bf a, v16bf b, v8f c) {
  return __builtin_amdgcn_wmma_f32_16x16x32_bf16(false, a, false, b, (short)0,
                                                 c, false, false);
}

__device__ __forceinline__ float sigmoidf_(float x) {
  return 1.f / (1.f + __expf(-x));
}

// ------------------------------- one-time prep -----------------------------

__global__ void k_cvt(const float* __restrict__ src, bf16* __restrict__ dst,
                      size_t n) {
  size_t i = (size_t)blockIdx.x * blockDim.x + threadIdx.x;
  if (i < n) dst[i] = (bf16)src[i];
}

// W_ih0 is [3H, E+H]; extract+convert the first E columns (emb part).
__global__ void k_cvt_w0(const float* __restrict__ W, bf16* __restrict__ dst) {
  size_t i = (size_t)blockIdx.x * blockDim.x + threadIdx.x;
  if (i >= (size_t)GG * EE) return;
  int r = (int)(i / EE), c = (int)(i % EE);
  dst[i] = (bf16)W[(size_t)r * (EE + HH) + c];
}

// Gather embeddings -> embs_bf [row=t*16+b, E]; also fill the static slices
// of pred [row, PREDW]: cols [H, H+LAT) = sem_z(b), cols [H+LAT, PREDW) = emb.
__global__ void k_embed(const int* __restrict__ seq,
                        const float* __restrict__ emb_table,
                        const float* __restrict__ sem_z,
                        bf16* __restrict__ embs_bf, bf16* __restrict__ pred_bf) {
  int r = blockIdx.x;             // 0..2047
  int b = r & 15, t = r >> 4;
  int tok = seq[b * TT + t];
  const float* e = emb_table + (size_t)tok * EE;
  bf16* erow = embs_bf + (size_t)r * EE;
  bf16* prow = pred_bf + (size_t)r * PREDW;
  for (int c = threadIdx.x; c < EE; c += blockDim.x) {
    bf16 v = (bf16)e[c];
    erow[c] = v;
    prow[2 * HH + c] = v;         // emb slice
  }
  const float* sz = sem_z + b * HH;
  for (int c = threadIdx.x; c < HH; c += blockDim.x)
    prow[HH + c] = (bf16)sz[c];   // sem_z slice
}

// syn_const[b,g] = syn_z[b,:] . W_ih0[g, E:E+H] + b_ih0[g]  (time-invariant)
__global__ void k_syn(const float* __restrict__ syn_z,
                      const float* __restrict__ W_ih0,
                      const float* __restrict__ b_ih0,
                      float* __restrict__ out) {
  int i = blockIdx.x * blockDim.x + threadIdx.x;
  if (i >= BB * GG) return;
  int b = i / GG, g = i % GG;
  const float4* x = reinterpret_cast<const float4*>(syn_z + (size_t)b * HH);
  const float4* w =
      reinterpret_cast<const float4*>(W_ih0 + (size_t)g * (EE + HH) + EE);
  float acc = 0.f;
  for (int k = 0; k < HH / 4; ++k) {
    float4 xa = x[k], wa = w[k];
    acc += xa.x * wa.x + xa.y * wa.y + xa.z * wa.z + xa.w * wa.w;
  }
  out[i] = acc + b_ih0[g];
}

__global__ void k_init_state(const float* __restrict__ hidden,
                             float* __restrict__ h0f, bf16* __restrict__ h0b,
                             float* __restrict__ h1f, bf16* __restrict__ h1b) {
  int i = blockIdx.x * blockDim.x + threadIdx.x;
  if (i >= BB * HH) return;
  float a = hidden[i];
  float c = hidden[BB * HH + i];
  h0f[i] = a; h0b[i] = (bf16)a;
  h1f[i] = c; h1b[i] = (bf16)c;
}

// ------------------------------ generic WMMA GEMM --------------------------
// C[M,N](f32) = A[M,K](bf16) @ B[N,K]^T (+bias). grid=(N/64, M/16), block=128.
// Used for gi_emb (weight is 3 MB bf16 -> L2-resident across M-tiles).
__global__ void __launch_bounds__(128)
k_gemm(const bf16* __restrict__ A, int lda, const bf16* __restrict__ B,
       int ldb, const float* __restrict__ bias, float* __restrict__ C, int ldc,
       int K) {
  const int lane = threadIdx.x & 31;
  const int wave = threadIdx.x >> 5;
  const int hi = lane >> 4, lr = lane & 15;
  const int m0 = blockIdx.y * 16;
  const int n0 = blockIdx.x * 64 + wave * 16;
  const bf16* arow = A + (size_t)(m0 + lr) * lda;
  const bf16* brow = B + (size_t)(n0 + lr) * ldb;
  v8f acc = {};
  for (int k0 = 0; k0 < K; k0 += 32) {
    v16bf a = load_a_frag(arow, k0, hi);
    v16bf b = load_b_frag(brow, k0, hi);
    acc = wmma_bf16(a, b, acc);
  }
  float bs = bias ? bias[n0 + lr] : 0.f;
  // C/D layout: element v -> row m0 + v + 8*hi, col n0 + lr.
  float* cp = C + (size_t)(m0 + 8 * hi) * ldc + (n0 + lr);
#pragma unroll
  for (int v = 0; v < 8; ++v) cp[(size_t)v * ldc] = acc[v] + bs;
}

// ------------------------- output projection GEMM --------------------------
// logits[2048, 32000] = pred(bf16) @ W_out(f32)^T + b_out.
// Block: 128 thr (4 waves). Stripe = 64 cols (16/wave). M-chunk = 256 rows
// (16 tiles, acc = 128 VGPRs/lane). K chunked by 64: the f32 W chunk is
// async-copied global->LDS (GLOBAL_LOAD_ASYNC_TO_LDS_B128, ASYNCcnt), then
// each B fragment is built once (ds_load_b128 + cvt) and reused 16x.
#define OP_NC 64    // cols per block
#define OP_KC 64    // k per LDS stage
#define OP_MT 16    // 16x16 M-tiles per block (256 rows)
#define OP_RSF 68   // padded f32 LDS row stride (272 B: 16B-aligned; 68-dword
                    // stride -> 16 fragment lanes hit distinct banks)

__global__ void __launch_bounds__(128)
k_outproj(const bf16* __restrict__ A, const float* __restrict__ W,
          const float* __restrict__ bias, float* __restrict__ C) {
  __shared__ float wlds[OP_NC * OP_RSF];  // 17408 B
  const int tid = threadIdx.x;
  const int lane = tid & 31, wave = tid >> 5;
  const int hi = lane >> 4, lr = lane & 15;
  const int n0 = blockIdx.x * OP_NC;         // global col base of stripe
  const int m0 = blockIdx.y * (OP_MT * 16);  // global row base
  const int nl = wave * 16 + lr;             // local col (fragment lane's n)

  v8f zf = {};
  v8f acc[OP_MT];
#pragma unroll
  for (int i = 0; i < OP_MT; ++i) acc[i] = zf;

  // Async staging assignment: thread covers W row sr = tid/2 and k-half
  // ck = (tid&1)*32 floats; 8 x 16B async copies per thread per chunk.
  const int sr = tid >> 1;
  const int ck = (tid & 1) * 32;
  unsigned long long gbase =
      (unsigned long long)(const void*)(W + (size_t)(n0 + sr) * PREDW + ck);
  unsigned lbase = (unsigned)(unsigned long long)(const void*)&wlds[sr * OP_RSF + ck];

  for (int kc = 0; kc < PREDW; kc += OP_KC) {
    __syncthreads();  // previous chunk's LDS reads complete
    {
      unsigned long long g = gbase + (unsigned long long)kc * 4u;
#pragma unroll
      for (int i = 0; i < 8; ++i) {
        asm volatile("global_load_async_to_lds_b128 %0, %1, off"
                     :
                     : "v"(lbase + (unsigned)(i * 16)),
                       "v"(g + (unsigned long long)(i * 16))
                     : "memory");
      }
      asm volatile("s_wait_asynccnt 0x0" ::: "memory");
    }
    __syncthreads();  // whole chunk staged by all waves
#pragma unroll
    for (int sub = 0; sub < 2; ++sub) {
      // B fragment: element e -> k = kc + sub*32 + 16*hi + e; built from
      // 16 consecutive f32 in LDS (4x ds_load_b128), converted to bf16.
      Frag bf_;
      const float* bp = &wlds[nl * OP_RSF + sub * 32 + 16 * hi];
#pragma unroll
      for (int i = 0; i < 4; ++i) {
        float4 x = reinterpret_cast<const float4*>(bp)[i];
        bf_.h[4 * i + 0] = (bf16)x.x;
        bf_.h[4 * i + 1] = (bf16)x.y;
        bf_.h[4 * i + 2] = (bf16)x.z;
        bf_.h[4 * i + 3] = (bf16)x.w;
      }
      const int ka = kc + sub * 32;
      // 16 M-tiles share this B fragment; group A loads by 4 so the compiler
      // can overlap loads with WMMA via partial s_wait_loadcnt.
#pragma unroll
      for (int mg = 0; mg < OP_MT; mg += 4) {
        v16bf a0 = load_a_frag(A + (size_t)(m0 + (mg + 0) * 16 + lr) * PREDW, ka, hi);
        v16bf a1 = load_a_frag(A + (size_t)(m0 + (mg + 1) * 16 + lr) * PREDW, ka, hi);
        v16bf a2 = load_a_frag(A + (size_t)(m0 + (mg + 2) * 16 + lr) * PREDW, ka, hi);
        v16bf a3 = load_a_frag(A + (size_t)(m0 + (mg + 3) * 16 + lr) * PREDW, ka, hi);
        acc[mg + 0] = wmma_bf16(a0, bf_.v, acc[mg + 0]);
        acc[mg + 1] = wmma_bf16(a1, bf_.v, acc[mg + 1]);
        acc[mg + 2] = wmma_bf16(a2, bf_.v, acc[mg + 2]);
        acc[mg + 3] = wmma_bf16(a3, bf_.v, acc[mg + 3]);
      }
    }
  }
  const float bs = bias[n0 + nl];
#pragma unroll
  for (int mt = 0; mt < OP_MT; ++mt) {
    float* cp = C + (size_t)(m0 + mt * 16 + 8 * hi) * VV + (n0 + nl);
#pragma unroll
    for (int v = 0; v < 8; ++v) cp[(size_t)v * VV] = acc[mt][v] + bs;
  }
}

// ------------------------------ GRU step kernels ---------------------------
// Layer 0: gh = h0_prev @ W_hh0^T ; gates fused with (gi_emb + syn_const).
// grid = 16 blocks x 128 thr = 64 waves; wave owns a 16-wide h-unit tile.
__global__ void __launch_bounds__(128)
k_stepA(const bf16* __restrict__ h0b_prev, const float* __restrict__ h0f_prev,
        const bf16* __restrict__ Whh0, const float* __restrict__ gi_emb,
        const float* __restrict__ syn_c, const float* __restrict__ b_hh0,
        float* __restrict__ h0f_new, bf16* __restrict__ h0b_new, int t) {
  const int lane = threadIdx.x & 31, wave = threadIdx.x >> 5;
  const int hi = lane >> 4, lr = lane & 15;
  const int nh = (blockIdx.x * 4 + wave) * 16;  // h-unit tile base
  const bf16* arow = h0b_prev + (size_t)lr * HH;
  const bf16* brR = Whh0 + (size_t)(nh + lr) * HH;
  const bf16* brZ = Whh0 + (size_t)(HH + nh + lr) * HH;
  const bf16* brN = Whh0 + (size_t)(2 * HH + nh + lr) * HH;
  v8f aR = {}, aZ = {}, aN = {};
  for (int k0 = 0; k0 < HH; k0 += 32) {
    v16bf a = load_a_frag(arow, k0, hi);
    aR = wmma_bf16(a, load_b_frag(brR, k0, hi), aR);
    aZ = wmma_bf16(a, load_b_frag(brZ, k0, hi), aZ);
    aN = wmma_bf16(a, load_b_frag(brN, k0, hi), aN);
  }
  const int j = nh + lr;
  const float bhR = b_hh0[j], bhZ = b_hh0[HH + j], bhN = b_hh0[2 * HH + j];
#pragma unroll
  for (int v = 0; v < 8; ++v) {
    int m = v + 8 * hi;  // batch row
    const float* gi = gi_emb + (size_t)(t * BB + m) * GG;
    const float* sc = syn_c + (size_t)m * GG;
    float ir = gi[j] + sc[j];
    float iz = gi[HH + j] + sc[HH + j];
    float in_ = gi[2 * HH + j] + sc[2 * HH + j];
    float r = sigmoidf_(ir + aR[v] + bhR);
    float z = sigmoidf_(iz + aZ[v] + bhZ);
    float n = tanhf(in_ + r * (aN[v] + bhN));
    float hp = h0f_prev[m * HH + j];
    float ho = (1.f - z) * n + z * hp;
    h0f_new[m * HH + j] = ho;
    h0b_new[m * HH + j] = (bf16)ho;
  }
}

// Layer 1: gi = h0 @ W_ih1^T, gh = h1_prev @ W_hh1^T; writes h1 state and the
// h1 slice of the pred matrix for the batched output projection.
__global__ void __launch_bounds__(128)
k_stepB(const bf16* __restrict__ h0b, const bf16* __restrict__ h1b_prev,
        const float* __restrict__ h1f_prev, const bf16* __restrict__ Wih1,
        const bf16* __restrict__ Whh1, const float* __restrict__ b_ih1,
        const float* __restrict__ b_hh1, float* __restrict__ h1f_new,
        bf16* __restrict__ h1b_new, bf16* __restrict__ pred_bf, int t) {
  const int lane = threadIdx.x & 31, wave = threadIdx.x >> 5;
  const int hi = lane >> 4, lr = lane & 15;
  const int nh = (blockIdx.x * 4 + wave) * 16;
  const bf16* a0row = h0b + (size_t)lr * HH;
  const bf16* a1row = h1b_prev + (size_t)lr * HH;
  const bf16* biR = Wih1 + (size_t)(nh + lr) * HH;
  const bf16* biZ = Wih1 + (size_t)(HH + nh + lr) * HH;
  const bf16* biN = Wih1 + (size_t)(2 * HH + nh + lr) * HH;
  const bf16* bhRp = Whh1 + (size_t)(nh + lr) * HH;
  const bf16* bhZp = Whh1 + (size_t)(HH + nh + lr) * HH;
  const bf16* bhNp = Whh1 + (size_t)(2 * HH + nh + lr) * HH;
  v8f iR = {}, iZ = {}, iN = {}, hR = {}, hZ = {}, hN = {};
  for (int k0 = 0; k0 < HH; k0 += 32) {
    v16bf a0 = load_a_frag(a0row, k0, hi);
    v16bf a1 = load_a_frag(a1row, k0, hi);
    iR = wmma_bf16(a0, load_b_frag(biR, k0, hi), iR);
    iZ = wmma_bf16(a0, load_b_frag(biZ, k0, hi), iZ);
    iN = wmma_bf16(a0, load_b_frag(biN, k0, hi), iN);
    hR = wmma_bf16(a1, load_b_frag(bhRp, k0, hi), hR);
    hZ = wmma_bf16(a1, load_b_frag(bhZp, k0, hi), hZ);
    hN = wmma_bf16(a1, load_b_frag(bhNp, k0, hi), hN);
  }
  const int j = nh + lr;
  const float biRj = b_ih1[j], biZj = b_ih1[HH + j], biNj = b_ih1[2 * HH + j];
  const float bhRj = b_hh1[j], bhZj = b_hh1[HH + j], bhNj = b_hh1[2 * HH + j];
#pragma unroll
  for (int v = 0; v < 8; ++v) {
    int m = v + 8 * hi;
    float r = sigmoidf_(iR[v] + biRj + hR[v] + bhRj);
    float z = sigmoidf_(iZ[v] + biZj + hZ[v] + bhZj);
    float n = tanhf(iN[v] + biNj + r * (hN[v] + bhNj));
    float hp = h1f_prev[m * HH + j];
    float ho = (1.f - z) * n + z * hp;
    h1f_new[m * HH + j] = ho;
    h1b_new[m * HH + j] = (bf16)ho;
    pred_bf[(size_t)(t * BB + m) * PREDW + j] = (bf16)ho;  // h1 slice of pred
  }
}

// ------------------------------- epilogues ---------------------------------

__global__ void k_final(const float* __restrict__ h0f,
                        const float* __restrict__ h1f,
                        float* __restrict__ out_hidden,
                        float* __restrict__ out_output) {
  int i = blockIdx.x * blockDim.x + threadIdx.x;
  if (i >= BB * HH) return;
  out_hidden[i] = h0f[i];
  out_hidden[BB * HH + i] = h1f[i];
  out_output[i] = h1f[i];
}

__global__ void k_rowstats(const float* __restrict__ logits,
                           float* __restrict__ lse) {
  int r = blockIdx.x;
  const float* row = logits + (size_t)r * VV;
  __shared__ float sm[256];
  float m = -1e30f;
  for (int i = threadIdx.x; i < VV; i += 256) m = fmaxf(m, row[i]);
  sm[threadIdx.x] = m;
  __syncthreads();
  for (int s = 128; s > 0; s >>= 1) {
    if (threadIdx.x < s) sm[threadIdx.x] = fmaxf(sm[threadIdx.x], sm[threadIdx.x + s]);
    __syncthreads();
  }
  m = sm[0];
  __syncthreads();
  float sum = 0.f;
  for (int i = threadIdx.x; i < VV; i += 256) sum += __expf(row[i] - m);
  sm[threadIdx.x] = sum;
  __syncthreads();
  for (int s = 128; s > 0; s >>= 1) {
    if (threadIdx.x < s) sm[threadIdx.x] += sm[threadIdx.x + s];
    __syncthreads();
  }
  if (threadIdx.x == 0) lse[r] = m + __logf(sm[0]);
}

__global__ void k_scores(const float* __restrict__ logits,
                         const float* __restrict__ lse,
                         float* __restrict__ scores, size_t total) {
  size_t i = (size_t)blockIdx.x * blockDim.x + threadIdx.x;
  if (i < total) scores[i] = logits[i] - lse[i / VV];
}

// --------------------------------- launcher --------------------------------

extern "C" void kernel_launch(void* const* d_in, const int* in_sizes, int n_in,
                              void* d_out, int out_size, void* d_ws,
                              size_t ws_size, hipStream_t stream) {
  (void)in_sizes; (void)n_in; (void)out_size; (void)ws_size;
  const int*   seq    = (const int*)d_in[0];
  // d_in[1] (prev_output) unused: input_feeding=False in reference.
  const float* sem_z  = (const float*)d_in[2];
  const float* syn_z  = (const float*)d_in[3];
  const float* hidden = (const float*)d_in[4];
  const float* emb_t  = (const float*)d_in[5];
  const float* W_ih0  = (const float*)d_in[6];
  const float* W_hh0  = (const float*)d_in[7];
  const float* b_ih0  = (const float*)d_in[8];
  const float* b_hh0  = (const float*)d_in[9];
  const float* W_ih1  = (const float*)d_in[10];
  const float* W_hh1  = (const float*)d_in[11];
  const float* b_ih1  = (const float*)d_in[12];
  const float* b_hh1  = (const float*)d_in[13];
  const float* W_out  = (const float*)d_in[14];
  const float* b_out  = (const float*)d_in[15];

  // d_out layout: scores | hidden_f | output | logits (flat, f32)
  float* outp = (float*)d_out;
  const size_t SCORES_OFF = 0;
  const size_t HIDDEN_OFF = (size_t)TT * BB * VV;  // 65,536,000
  const size_t OUTPUT_OFF = HIDDEN_OFF + 2 * BB * HH;
  const size_t LOGITS_OFF = OUTPUT_OFF + BB * HH;
  float* scores = outp + SCORES_OFF;
  float* logits = outp + LOGITS_OFF;

  // Workspace carve-out (~61 MB), 256B aligned.
  char* w = (char*)d_ws;
  auto carve = [&](size_t bytes) {
    char* p = w;
    w += (bytes + 255) & ~(size_t)255;
    return p;
  };
  bf16*  Wih0e_bf = (bf16*)carve((size_t)GG * EE * 2);
  bf16*  Whh0_bf  = (bf16*)carve((size_t)GG * HH * 2);
  bf16*  Wih1_bf  = (bf16*)carve((size_t)GG * HH * 2);
  bf16*  Whh1_bf  = (bf16*)carve((size_t)GG * HH * 2);
  bf16*  embs_bf  = (bf16*)carve((size_t)ROWS * EE * 2);
  bf16*  pred_bf  = (bf16*)carve((size_t)ROWS * PREDW * 2);
  float* gi_emb   = (float*)carve((size_t)ROWS * GG * 4);
  float* syn_c    = (float*)carve((size_t)BB * GG * 4);
  float* h0f[2], *h1f[2];
  bf16*  h0b[2], *h1b[2];
  for (int i = 0; i < 2; ++i) {
    h0f[i] = (float*)carve((size_t)BB * HH * 4);
    h1f[i] = (float*)carve((size_t)BB * HH * 4);
    h0b[i] = (bf16*)carve((size_t)BB * HH * 2);
    h1b[i] = (bf16*)carve((size_t)BB * HH * 2);
  }
  float* lse = (float*)carve((size_t)ROWS * 4);

  // --- one-time prep ---
  {
    size_t n = (size_t)GG * EE;
    k_cvt_w0<<<(int)((n + 255) / 256), 256, 0, stream>>>(W_ih0, Wih0e_bf);
    n = (size_t)GG * HH;
    int g = (int)((n + 255) / 256);
    k_cvt<<<g, 256, 0, stream>>>(W_hh0, Whh0_bf, n);
    k_cvt<<<g, 256, 0, stream>>>(W_ih1, Wih1_bf, n);
    k_cvt<<<g, 256, 0, stream>>>(W_hh1, Whh1_bf, n);
  }
  k_embed<<<ROWS, 256, 0, stream>>>(seq, emb_t, sem_z, embs_bf, pred_bf);
  k_syn<<<(BB * GG + 255) / 256, 256, 0, stream>>>(syn_z, W_ih0, b_ih0, syn_c);
  k_init_state<<<(BB * HH + 255) / 256, 256, 0, stream>>>(hidden, h0f[0],
                                                          h0b[0], h1f[0], h1b[0]);

  // gi_emb[row, 3H] = embs @ W_ih0[:, :E]^T   (all timesteps batched)
  {
    dim3 grid(GG / 64, ROWS / 16);
    k_gemm<<<grid, 128, 0, stream>>>(embs_bf, EE, Wih0e_bf, EE, nullptr,
                                     gi_emb, GG, EE);
  }

  // --- sequential recurrence: 128 steps, ping-pong state ---
  for (int t = 0; t < TT; ++t) {
    int cur = t & 1, nxt = cur ^ 1;
    k_stepA<<<16, 128, 0, stream>>>(h0b[cur], h0f[cur], Whh0_bf, gi_emb, syn_c,
                                    b_hh0, h0f[nxt], h0b[nxt], t);
    k_stepB<<<16, 128, 0, stream>>>(h0b[nxt], h1b[cur], h1f[cur], Wih1_bf,
                                    Whh1_bf, b_ih1, b_hh1, h1f[nxt], h1b[nxt],
                                    pred_bf, t);
  }
  const int fin = TT & 1;  // final state buffer index (0 for T=128)
  k_final<<<(BB * HH + 255) / 256, 256, 0, stream>>>(
      h0f[fin], h1f[fin], outp + HIDDEN_OFF, outp + OUTPUT_OFF);

  // --- batched output projection with async-LDS-staged, M-reused W_out ---
  {
    dim3 grid(VV / OP_NC, ROWS / (OP_MT * 16));  // (500, 8)
    k_outproj<<<grid, 128, 0, stream>>>(pred_bf, W_out, b_out, logits);
  }

  // --- log_softmax ---
  k_rowstats<<<ROWS, 256, 0, stream>>>(logits, lse);
  {
    size_t total = (size_t)ROWS * VV;
    k_scores<<<(int)((total + 255) / 256), 256, 0, stream>>>(logits, lse,
                                                             scores, total);
  }
}
